// GRU_80126909874387
// MI455X (gfx1250) — matrix operand order repaired and checked by reading, not compiled, not run on hardware
//
#include <hip/hip_runtime.h>
#include <hip/hip_bf16.h>
#include <math.h>

typedef __attribute__((ext_vector_type(16))) _Float16 v16h;
typedef __attribute__((ext_vector_type(8)))  float    v8f;

#define TT    512          // timesteps (axis 0 of emb)
#define EMB   128
#define HID   256
#define G3    (3*HID)      // 768

// ---------------------------------------------------------------------------
// K1: gather embeddings for batch column n = 511 only (rest of batch is dead)
// ---------------------------------------------------------------------------
__global__ void k_gather(const int* __restrict__ x,
                         const float* __restrict__ tab,
                         float* __restrict__ E) {
    int t = blockIdx.x;
    int d = threadIdx.x;
    int tok = x[t * 512 + 511];
    E[t * EMB + d] = tab[tok * EMB + d];
}

// ---------------------------------------------------------------------------
// K2: XI = A (MxK, row-major) @ W^T (W is NxK row-major) + bias, via WMMA.
// One wave per 16x16 output tile, f16 inputs, f32 accumulate.
// ---------------------------------------------------------------------------
__global__ void __launch_bounds__(128) k_gemm_xi(
        const float* __restrict__ A,     // M x K
        const float* __restrict__ W,     // N x K  (we use W^T)
        const float* __restrict__ bias,  // N
        float* __restrict__ out,         // M x N
        int M, int N, int K) {
    int wave = (blockIdx.x * blockDim.x + threadIdx.x) >> 5;
    int lane = threadIdx.x & 31;
    int tilesN = N >> 4;
    int tm = (wave / tilesN) << 4;
    int tn = (wave % tilesN) << 4;

    int l16 = lane & 15;
    int hi  = lane >> 4;                 // 0: lanes 0-15, 1: lanes 16-31

    const float* arow = A + (tm + l16) * K;   // A row for this lane (M = lane%16)
    const float* wrow = W + (tn + l16) * K;   // W row for this lane (N = lane%16)

    v8f acc = {};
    for (int k0 = 0; k0 < K; k0 += 32) {
        v16h a, b;
#pragma unroll
        for (int e = 0; e < 16; ++e) {
            // A 16x32 layout: halfs 0-7 -> K base+0..7 (lo lanes) / +8..15 (hi),
            //                 halfs 8-15 -> K +16..23 (lo) / +24..31 (hi)
            int ka = (e < 8 ? e : e + 8) + hi * 8;
            a[e] = (_Float16)arow[k0 + ka];
            // B 32x16 layout: same K order as A (0-7/16-23 lo lanes, 8-15/24-31 hi)
            int kb = (e < 8 ? e : e + 8) + hi * 8;
            b[e] = (_Float16)wrow[k0 + kb];
        }
        acc = __builtin_amdgcn_wmma_f32_16x16x32_f16(
                  false, a, false, b, (short)0, acc, false, false);
    }
    // D layout: VGPR i holds M = i (lanes 0-15) / i+8 (lanes 16-31), N = lane%16
#pragma unroll
    for (int i = 0; i < 8; ++i) {
        int row = tm + i + hi * 8;
        int col = tn + l16;
        out[row * N + col] = acc[i] + bias[col];
    }
}

// ---------------------------------------------------------------------------
// K3: sequential GRU layer. One workgroup, 768 threads (24 wave32s).
// Thread j computes gh[j] = h . Whh[j,:] (float4-streamed from L2/WGP$),
// threads 0..255 then apply the gate nonlinearity and update h.
// ---------------------------------------------------------------------------
__global__ void __launch_bounds__(G3, 1) k_gru_seq(
        const float* __restrict__ XI,    // T x 768 (precomputed x-gates)
        const float* __restrict__ Whh,   // 768 x 256 row-major
        const float* __restrict__ bhh,   // 768
        float* __restrict__ Hout) {      // T x 256 hidden trajectory
    __shared__ __align__(16) float h[HID];
    __shared__ float gh[G3];

    int j = threadIdx.x;                 // 0..767
    if (j < HID) h[j] = 0.0f;
    float bj = bhh[j];
    const float4* wrow = (const float4*)(Whh + j * HID);  // 64 float4 per row
    __syncthreads();

    for (int t = 0; t < TT; ++t) {
        const float4* h4 = (const float4*)h;
        float acc = 0.0f;
#pragma unroll 8
        for (int k = 0; k < HID / 4; ++k) {
            float4 w = wrow[k];
            float4 v = h4[k];            // LDS broadcast
            acc += w.x * v.x + w.y * v.y + w.z * v.z + w.w * v.w;
        }
        gh[j] = acc + bj;
        __syncthreads();                 // all reads of h_{t-1} done, gh visible

        if (j < HID) {
            const float* xi = XI + t * G3;
            float r  = 1.0f / (1.0f + __expf(-(xi[j]           + gh[j])));
            float z  = 1.0f / (1.0f + __expf(-(xi[HID + j]     + gh[HID + j])));
            float nn = tanhf(xi[2 * HID + j] + r * gh[2 * HID + j]);
            float hn = (1.0f - z) * nn + z * h[j];
            h[j] = hn;
            Hout[t * HID + j] = hn;
        }
        __syncthreads();                 // h_t visible before next step's reads
    }
}

// ---------------------------------------------------------------------------
// K4: logits = H2 @ W_out^T + b_out   (512 x 2, trivial)
// ---------------------------------------------------------------------------
__global__ void k_logits(const float* __restrict__ H2,
                         const float* __restrict__ Wout,
                         const float* __restrict__ bout,
                         float* __restrict__ out) {
    int i = blockIdx.x * blockDim.x + threadIdx.x;  // 0..1023
    int t = i >> 1;
    int c = i & 1;
    float acc = bout[c];
#pragma unroll 4
    for (int k = 0; k < HID; ++k)
        acc += H2[t * HID + k] * Wout[c * HID + k];
    out[t * 2 + c] = acc;
}

// ---------------------------------------------------------------------------
extern "C" void kernel_launch(void* const* d_in, const int* in_sizes, int n_in,
                              void* d_out, int out_size, void* d_ws, size_t ws_size,
                              hipStream_t stream) {
    const int*   x    = (const int*)  d_in[0];
    const float* tab  = (const float*)d_in[1];
    const float* Wih0 = (const float*)d_in[2];
    const float* Whh0 = (const float*)d_in[3];
    const float* bih0 = (const float*)d_in[4];
    const float* bhh0 = (const float*)d_in[5];
    const float* Wih1 = (const float*)d_in[6];
    const float* Whh1 = (const float*)d_in[7];
    const float* bih1 = (const float*)d_in[8];
    const float* bhh1 = (const float*)d_in[9];
    const float* Wout = (const float*)d_in[10];
    const float* bout = (const float*)d_in[11];
    float* out = (float*)d_out;

    float* ws = (float*)d_ws;
    float* E  = ws;                  // 512*128
    float* XI = E  + TT * EMB;       // 512*768 (reused for both layers)
    float* H1 = XI + TT * G3;        // 512*256
    float* H2 = H1 + TT * HID;       // 512*256

    // 1) gather the single live batch column
    k_gather<<<TT, EMB, 0, stream>>>(x, tab, E);

    // 2) layer-1 input gates (WMMA GEMM, K=128)
    int waves = (TT / 16) * (G3 / 16);          // 1536 tiles
    k_gemm_xi<<<waves / 4, 128, 0, stream>>>(E, Wih0, bih0, XI, TT, G3, EMB);

    // 3) layer-1 recurrence (sequential, single WGP)
    k_gru_seq<<<1, G3, 0, stream>>>(XI, Whh0, bhh0, H1);

    // 4) layer-2 input gates (WMMA GEMM, K=256)
    k_gemm_xi<<<waves / 4, 128, 0, stream>>>(H1, Wih1, bih1, XI, TT, G3, HID);

    // 5) layer-2 recurrence
    k_gru_seq<<<1, G3, 0, stream>>>(XI, Whh1, bhh1, H2);

    // 6) output projection
    k_logits<<<(TT * 2) / 256, 256, 0, stream>>>(H2, Wout, bout, out);
}
